// AttnBlock_25847113188076
// MI455X (gfx1250) — matrix operand order repaired and checked by reading, not compiled, not run on hardware
//
#include <hip/hip_runtime.h>
#include <hip/hip_bf16.h>

typedef __attribute__((ext_vector_type(16))) _Float16 v16h;
typedef __attribute__((ext_vector_type(8)))  _Float16 v8h;
typedef __attribute__((ext_vector_type(8)))  float    v8f;

#define HEADS 12
#define DH    64
#define DIM   768
#define NSEQ  1025
#define BS    8
#define MREAL (BS * NSEQ)   /* 8200 */
#define MPAD  8256          /* 129 * 64 */
#define NQKV  (3 * DIM)     /* 2304 */
#define EPSF  1e-5f

// ---------------------------------------------------------------------------
// WMMA helpers (CDNA5 wave32, v_wmma_f32_16x16x32_f16)
// ---------------------------------------------------------------------------
__device__ __forceinline__ v8f wmma16(v16h a, v16h b, v8f c) {
  return __builtin_amdgcn_wmma_f32_16x16x32_f16(
      /*neg_a=*/false, a, /*neg_b=*/false, b,
      /*c_mod=*/(short)0, c, /*reuse_a=*/false, /*reuse_b=*/false);
}

// A-fragment (16x32 f16): row = lane&15; lanes>=16 take K offset +16;
// 16 consecutive K values per lane.
__device__ __forceinline__ v16h load_a16(const _Float16* p, int stride, int lane) {
  const int r = lane & 15;
  const _Float16* row = p + r * stride + ((lane >> 4) << 4);
  v16h f;
#pragma unroll
  for (int j = 0; j < 8; ++j) {
    const int k0 = 2 * j;
    f[2 * j]     = row[k0];
    f[2 * j + 1] = row[k0 + 1];
  }
  return f;
}

// B-fragment (32x16 f16): col n = lane&15, K base = (lane>>4)*16,
// 16 consecutive K values per lane. p must point at [n][kbase] of a
// [n][k]-layout (i.e. transposed) tile.
__device__ __forceinline__ v16h load_b16(const _Float16* p) {
  v16h f;
#pragma unroll
  for (int i = 0; i < 16; ++i) f[i] = p[i];
  return f;
}

// reductions across the 16 lanes of a C-fragment row (xor masks stay in-half)
__device__ __forceinline__ float redmax16(float v) {
  v = fmaxf(v, __shfl_xor(v, 1, 32));
  v = fmaxf(v, __shfl_xor(v, 2, 32));
  v = fmaxf(v, __shfl_xor(v, 4, 32));
  v = fmaxf(v, __shfl_xor(v, 8, 32));
  return v;
}
__device__ __forceinline__ float redsum16(float v) {
  v += __shfl_xor(v, 1, 32);
  v += __shfl_xor(v, 2, 32);
  v += __shfl_xor(v, 4, 32);
  v += __shfl_xor(v, 8, 32);
  return v;
}

// ---------------------------------------------------------------------------
// Conversion / init kernels
// ---------------------------------------------------------------------------
__global__ void cvt_pad_f32_f16(const float* __restrict__ s, _Float16* __restrict__ d,
                                long rows_real, long total, int cols) {
  long i = (long)blockIdx.x * blockDim.x + threadIdx.x;
  if (i >= total) return;
  long r = i / cols;
  d[i] = (r < rows_real) ? (_Float16)s[i] : (_Float16)0.f;
}

__global__ void cvt_f32_f16(const float* __restrict__ s, _Float16* __restrict__ d, long n) {
  long i = (long)blockIdx.x * blockDim.x + threadIdx.x;
  if (i < n) d[i] = (_Float16)s[i];
}

__global__ void zero_f16(_Float16* __restrict__ d, long n) {
  long i = (long)blockIdx.x * blockDim.x + threadIdx.x;
  if (i < n) d[i] = (_Float16)0.f;
}

// ---------------------------------------------------------------------------
// WMMA GEMM: C[M,N] = A[M,K] @ B[K,N] (+bias). A,B f16 row-major, f32 accum.
// Block tile 64x256, 8 waves (2x4), each wave 32x64 (8 C-frags), K step 32.
// Register double-buffered global->LDS staging + global_prefetch_b8 ahead.
// EPI 0: QKV epilogue -> scatter f16 into q/k/v [b,h,n,dh] layout.
// EPI 1: proj epilogue -> f32 into d_out.
// ---------------------------------------------------------------------------
template <int EPI>
__global__ __launch_bounds__(256) void gemm_f16(
    const _Float16* __restrict__ A, const _Float16* __restrict__ Bw,
    const float* __restrict__ bias,
    _Float16* __restrict__ qbuf, _Float16* __restrict__ kbuf, _Float16* __restrict__ vbuf,
    float* __restrict__ outF, int N, int K, int mtiles) {
  __shared__ _Float16 ldsA[64 * 34];     // [row][k], stride 34
  __shared__ _Float16 ldsBt[256 * 34];   // transposed: [n][k], stride 34

  const int tid  = threadIdx.x;
  const int lane = tid & 31, wave = tid >> 5;
  const int wm = wave & 1, wn = wave >> 1;
  const int mt = blockIdx.x % mtiles, nt = blockIdx.x / mtiles;
  const int m0 = mt * 64, n0 = nt * 256;
  const int ln = lane & 15, lh = lane >> 4;

  v8f C[2][4];
#pragma unroll
  for (int i = 0; i < 2; ++i)
#pragma unroll
    for (int j = 0; j < 4; ++j)
#pragma unroll
      for (int e = 0; e < 8; ++e) C[i][j][e] = 0.f;

  const int arow = tid >> 2, acg = (tid & 3) * 8;   // A: 64 rows x 32 k
  const int bkk = tid >> 3, bng = (tid & 7) * 32;   // B: 32 k x 256 n

  // prologue: stage first K-tile into registers
  v8h avr = *(const v8h*)(A + (size_t)(m0 + arow) * K + acg);
  v8h bvr[4];
#pragma unroll
  for (int i = 0; i < 4; ++i)
    bvr[i] = *(const v8h*)(Bw + (size_t)bkk * N + n0 + bng + 8 * i);

  for (int kt = 0; kt < K; kt += 32) {
    __syncthreads();
    {  // commit staged registers to LDS (A row-major, B transposed [n][k])
      _Float16* d = &ldsA[arow * 34 + acg];
#pragma unroll
      for (int i = 0; i < 8; ++i) d[i] = avr[i];
#pragma unroll
      for (int g = 0; g < 4; ++g)
#pragma unroll
        for (int i = 0; i < 8; ++i)
          ldsBt[(bng + g * 8 + i) * 34 + bkk] = bvr[g][i];
    }
    __syncthreads();

    // issue next tile's global loads (latency hidden under 8 WMMAs)
    if (kt + 32 < K) {
      avr = *(const v8h*)(A + (size_t)(m0 + arow) * K + (kt + 32) + acg);
#pragma unroll
      for (int i = 0; i < 4; ++i)
        bvr[i] = *(const v8h*)(Bw + (size_t)(kt + 32 + bkk) * N + n0 + bng + 8 * i);
    }
    if (kt + 64 < K) {  // gfx1250 global_prefetch_b8 two tiles ahead
      __builtin_prefetch((const void*)(A + (size_t)(m0 + arow) * K + (kt + 64) + acg), 0, 1);
      __builtin_prefetch((const void*)(Bw + (size_t)(kt + 64 + bkk) * N + n0 + bng), 0, 1);
    }

    v16h a0 = load_a16(&ldsA[(wm * 32) * 34], 34, lane);
    v16h a1 = load_a16(&ldsA[(wm * 32 + 16) * 34], 34, lane);
#pragma unroll
    for (int nf = 0; nf < 4; ++nf) {
      v16h b = load_b16(&ldsBt[(wn * 64 + nf * 16 + ln) * 34 + lh * 16]);
      C[0][nf] = wmma16(a0, b, C[0][nf]);
      C[1][nf] = wmma16(a1, b, C[1][nf]);
    }
  }

  // Epilogue. C-frag element r lives at (m = r + 8*(lane>>4), n = lane&15).
#pragma unroll
  for (int mf = 0; mf < 2; ++mf)
#pragma unroll
    for (int nf = 0; nf < 4; ++nf) {
      const int n = n0 + wn * 64 + nf * 16 + ln;
      const float bv = bias[n];
      if (EPI == 0) {
        const int region = n / DIM;                 // 0=q 1=k 2=v
        const int c = n - region * DIM;
        const int hh = c >> 6, d = c & 63;
        _Float16* dst = (region == 0) ? qbuf : ((region == 1) ? kbuf : vbuf);
#pragma unroll
        for (int r = 0; r < 8; ++r) {
          const int m = m0 + wm * 32 + mf * 16 + r + (lh << 3);
          if (m < MREAL) {
            const int bi = m / NSEQ;
            const int nn = m - bi * NSEQ;
            dst[(((size_t)bi * HEADS + hh) * NSEQ + nn) * DH + d] =
                (_Float16)(C[mf][nf][r] + bv);
          }
        }
      } else {
#pragma unroll
        for (int r = 0; r < 8; ++r) {
          const int m = m0 + wm * 32 + mf * 16 + r + (lh << 3);
          if (m < MREAL) outF[(size_t)m * N + n] = C[mf][nf][r] + bv;
        }
      }
    }
}

// ---------------------------------------------------------------------------
// In-place LayerNorm over dh=64 for q then k. One wave per row.
// ---------------------------------------------------------------------------
__global__ __launch_bounds__(256) void ln_qk(
    _Float16* __restrict__ qbuf, _Float16* __restrict__ kbuf,
    const float* __restrict__ gq, const float* __restrict__ bq,
    const float* __restrict__ gk, const float* __restrict__ bk) {
  const int lane = threadIdx.x & 31, wave = threadIdx.x >> 5;
  const long row = (long)blockIdx.x * 8 + wave;
  const long QR = (long)BS * HEADS * NSEQ;
  _Float16* p; const float *g, *be;
  if (row < QR) { p = qbuf + row * DH; g = gq; be = bq; }
  else          { p = kbuf + (row - QR) * DH; g = gk; be = bk; }
  const int d0 = lane * 2;
  float x0 = (float)p[d0], x1 = (float)p[d0 + 1];
  float s = x0 + x1, sq = x0 * x0 + x1 * x1;
#pragma unroll
  for (int m = 1; m <= 16; m <<= 1) {
    s  += __shfl_xor(s, m, 32);
    sq += __shfl_xor(sq, m, 32);
  }
  const float mean = s * (1.f / 64.f);
  const float var  = fmaxf(sq * (1.f / 64.f) - mean * mean, 0.f);
  const float rstd = rsqrtf(var + EPSF);
  p[d0]     = (_Float16)((x0 - mean) * rstd * g[d0] + be[d0]);
  p[d0 + 1] = (_Float16)((x1 - mean) * rstd * g[d0 + 1] + be[d0 + 1]);
}

// ---------------------------------------------------------------------------
// Flash attention. Block = 8 waves; wave owns 16 query rows (block: 128).
// Grid: (bs*heads) * ceil(1025/128)=9. Streams keys in 32-wide tiles.
// ---------------------------------------------------------------------------
__global__ __launch_bounds__(256) void flash_attn(
    const _Float16* __restrict__ qb, const _Float16* __restrict__ kb,
    const _Float16* __restrict__ vb, _Float16* __restrict__ ob) {
  __shared__ _Float16 ldsK[32 * 66];       // [key][dh], B-fragment orientation for S
  __shared__ _Float16 ldsV[64 * 34];       // transposed [dh][key] for PV
  __shared__ _Float16 ldsP[8 * 16 * 34];   // per-wave P re-layout buffer

  const int tid = threadIdx.x, lane = tid & 31, wave = tid >> 5;
  const int qt = blockIdx.x % 9, bh = blockIdx.x / 9;
  const int b = bh / HEADS, h = bh - b * HEADS;
  const _Float16* qh = qb + (size_t)bh * NSEQ * DH;
  const _Float16* kh = kb + (size_t)bh * NSEQ * DH;
  const _Float16* vh = vb + (size_t)bh * NSEQ * DH;
  const int row0 = qt * 128 + wave * 16;
  const int ln = lane & 15, lh = lane >> 4;

  // Q fragments straight from global (A-layout): row = row0 + (lane&15), clamped
  v16h qa[2];
#pragma unroll
  for (int ch = 0; ch < 2; ++ch) {
    int m = row0 + ln;
    if (m > NSEQ - 1) m = NSEQ - 1;
    const _Float16* row = qh + (size_t)m * DH + ch * 32 + lh * 16;
    v16h f;
#pragma unroll
    for (int j = 0; j < 8; ++j) {
      const int k0 = 2 * j;
      f[2 * j] = row[k0]; f[2 * j + 1] = row[k0 + 1];
    }
    qa[ch] = f;
  }

  float mrun[8], lrun[8];
  v8f O[4];
#pragma unroll
  for (int r = 0; r < 8; ++r) { mrun[r] = -1e30f; lrun[r] = 0.f; }
#pragma unroll
  for (int f = 0; f < 4; ++f)
#pragma unroll
    for (int e = 0; e < 8; ++e) O[f][e] = 0.f;

  const float SL = 0.125f * 1.44269504088896341f;  // scale * log2(e)
  const int skey = tid >> 3, sdg = (tid & 7) * 8;

  for (int kt = 0; kt < NSEQ; kt += 32) {
    __syncthreads();
    {  // cooperative stage of K (row-major) and V (transposed)
      const int kg = kt + skey;
      v8h kv, vv;
      if (kg < NSEQ) {
        kv = *(const v8h*)(kh + (size_t)kg * DH + sdg);
        vv = *(const v8h*)(vh + (size_t)kg * DH + sdg);
      } else {
#pragma unroll
        for (int i = 0; i < 8; ++i) { kv[i] = (_Float16)0.f; vv[i] = (_Float16)0.f; }
      }
      _Float16* dk = &ldsK[skey * 66 + sdg];
#pragma unroll
      for (int i = 0; i < 8; ++i) {
        dk[i] = kv[i];
        ldsV[(sdg + i) * 34 + skey] = vv[i];
      }
    }
    __syncthreads();

    // prefetch next key/value tile while computing this one
    if (kt + 32 < NSEQ) {
      int kg2 = kt + 32 + skey;
      if (kg2 > NSEQ - 1) kg2 = NSEQ - 1;
      __builtin_prefetch((const void*)(kh + (size_t)kg2 * DH + sdg), 0, 1);
      __builtin_prefetch((const void*)(vh + (size_t)kg2 * DH + sdg), 0, 1);
    }

    // S = Q @ K^T  (two 16-key fragments, contraction dh=64 in two chunks)
    v8f S0, S1;
#pragma unroll
    for (int e = 0; e < 8; ++e) { S0[e] = 0.f; S1[e] = 0.f; }
#pragma unroll
    for (int ch = 0; ch < 2; ++ch) {
      v16h bk0 = load_b16(&ldsK[ln * 66 + ch * 32 + lh * 16]);
      v16h bk1 = load_b16(&ldsK[(16 + ln) * 66 + ch * 32 + lh * 16]);
      S0 = wmma16(qa[ch], bk0, S0);
      S1 = wmma16(qa[ch], bk1, S1);
    }
    if (kt + ln >= NSEQ) {
#pragma unroll
      for (int r = 0; r < 8; ++r) S0[r] = -1e30f;
    }
    if (kt + 16 + ln >= NSEQ) {
#pragma unroll
      for (int r = 0; r < 8; ++r) S1[r] = -1e30f;
    }

    // online softmax (per C-fragment row; stats replicated across 16 lanes)
    float p0[8], p1[8];
#pragma unroll
    for (int r = 0; r < 8; ++r) {
      const float mx   = redmax16(fmaxf(S0[r], S1[r]));
      const float mnew = fmaxf(mrun[r], mx);
      const float cf   = exp2f((mrun[r] - mnew) * SL);
      const float e0   = exp2f((S0[r] - mnew) * SL);
      const float e1   = exp2f((S1[r] - mnew) * SL);
      lrun[r] = lrun[r] * cf + redsum16(e0 + e1);
      mrun[r] = mnew;
      p0[r] = e0; p1[r] = e1;
      O[0][r] *= cf; O[1][r] *= cf; O[2][r] *= cf; O[3][r] *= cf;
    }

    // C-layout P -> wave-private LDS -> A-fragment (in-order DS per wave)
    _Float16* pw = &ldsP[wave * 16 * 34];
#pragma unroll
    for (int r = 0; r < 8; ++r) {
      const int m = r + (lh << 3);
      pw[m * 34 + ln]      = (_Float16)p0[r];
      pw[m * 34 + 16 + ln] = (_Float16)p1[r];
    }
    v16h pa = load_a16(pw, 34, lane);

    // O += P @ V
#pragma unroll
    for (int nf = 0; nf < 4; ++nf) {
      v16h bv = load_b16(&ldsV[(nf * 16 + ln) * 34 + lh * 16]);
      O[nf] = wmma16(pa, bv, O[nf]);
    }
  }

  // normalize and store [b, n, h*64+d] f16
#pragma unroll
  for (int r = 0; r < 8; ++r) {
    const int m = row0 + r + (lh << 3);
    if (m < NSEQ) {
      const float inv = 1.f / lrun[r];
      _Float16* dst = ob + ((size_t)(b * NSEQ + m)) * DIM + h * DH;
#pragma unroll
      for (int nf = 0; nf < 4; ++nf)
        dst[nf * 16 + ln] = (_Float16)(O[nf][r] * inv);
    }
  }
}

// ---------------------------------------------------------------------------
// attn_map = mean over heads of softmax(q0·K^T/8)[1:]. One block per batch,
// deterministic LDS reductions (no FP atomics).
// ---------------------------------------------------------------------------
__global__ __launch_bounds__(256) void attn_map_kernel(
    const _Float16* __restrict__ qb, const _Float16* __restrict__ kb,
    float* __restrict__ dmap) {
  __shared__ float q0s[64];
  __shared__ float red[256];
  const int b = blockIdx.x, tid = threadIdx.x;
  float acc[5] = {0.f, 0.f, 0.f, 0.f, 0.f};

  for (int h = 0; h < HEADS; ++h) {
    const size_t base = ((size_t)(b * HEADS + h)) * NSEQ * DH;
    if (tid < 64) q0s[tid] = (float)qb[base + tid];
    __syncthreads();

    float s[5];
    float mx = -1e30f;
    int cnt = 0;
    for (int j = tid; j < NSEQ; j += 256) {
      const _Float16* krow = kb + base + (size_t)j * DH;
      float d = 0.f;
      for (int e = 0; e < DH; ++e) d += q0s[e] * (float)krow[e];
      s[cnt] = d * 0.125f;
      mx = fmaxf(mx, s[cnt]);
      ++cnt;
    }
    red[tid] = mx; __syncthreads();
    for (int off = 128; off > 0; off >>= 1) {
      if (tid < off) red[tid] = fmaxf(red[tid], red[tid + off]);
      __syncthreads();
    }
    mx = red[0]; __syncthreads();

    float e[5], lsum = 0.f;
    for (int i = 0; i < cnt; ++i) { e[i] = __expf(s[i] - mx); lsum += e[i]; }
    red[tid] = lsum; __syncthreads();
    for (int off = 128; off > 0; off >>= 1) {
      if (tid < off) red[tid] += red[tid + off];
      __syncthreads();
    }
    const float inv = 1.f / red[0];
    __syncthreads();
    for (int i = 0; i < cnt; ++i) {
      const int j = tid + i * 256;
      if (j >= 1) acc[i] += e[i] * inv * (1.f / 12.f);
    }
  }
#pragma unroll
  for (int i = 0; i < 5; ++i) {
    const int j = tid + i * 256;
    if (j >= 1 && j < NSEQ) dmap[(size_t)b * (NSEQ - 1) + (j - 1)] = acc[i];
  }
}

// ---------------------------------------------------------------------------
extern "C" void kernel_launch(void* const* d_in, const int* in_sizes, int n_in,
                              void* d_out, int out_size, void* d_ws, size_t ws_size,
                              hipStream_t stream) {
  (void)in_sizes; (void)n_in; (void)out_size; (void)ws_size;
  const float* x     = (const float*)d_in[0];
  const float* Wqkv  = (const float*)d_in[1];
  const float* bqkv  = (const float*)d_in[2];
  const float* Wproj = (const float*)d_in[3];
  const float* bproj = (const float*)d_in[4];
  const float* gq    = (const float*)d_in[5];
  const float* betaq = (const float*)d_in[6];
  const float* gk    = (const float*)d_in[7];
  const float* betak = (const float*)d_in[8];

  float* out  = (float*)d_out;
  float* dmap = out + (size_t)MREAL * DIM;

  char* ws = (char*)d_ws;
  size_t off = 0;
  auto alloc = [&](size_t bytes) {
    char* p = ws + off;
    off = (off + bytes + 255) & ~(size_t)255;
    return p;
  };
  _Float16* x_h     = (_Float16*)alloc((size_t)MPAD * DIM * 2);
  _Float16* wqkv_h  = (_Float16*)alloc((size_t)DIM * NQKV * 2);
  _Float16* wproj_h = (_Float16*)alloc((size_t)DIM * DIM * 2);
  _Float16* qbuf    = (_Float16*)alloc((size_t)BS * HEADS * NSEQ * DH * 2);
  _Float16* kbuf    = (_Float16*)alloc((size_t)BS * HEADS * NSEQ * DH * 2);
  _Float16* vbuf    = (_Float16*)alloc((size_t)BS * HEADS * NSEQ * DH * 2);
  _Float16* obuf    = (_Float16*)alloc((size_t)MPAD * DIM * 2);

  const long xtot = (long)MPAD * DIM;
  cvt_pad_f32_f16<<<(int)((xtot + 255) / 256), 256, 0, stream>>>(x, x_h, MREAL, xtot, DIM);
  const long w1 = (long)DIM * NQKV;
  cvt_f32_f16<<<(int)((w1 + 255) / 256), 256, 0, stream>>>(Wqkv, wqkv_h, w1);
  const long w2 = (long)DIM * DIM;
  cvt_f32_f16<<<(int)((w2 + 255) / 256), 256, 0, stream>>>(Wproj, wproj_h, w2);
  const long zn = (long)(MPAD - MREAL) * DIM;
  zero_f16<<<(int)((zn + 255) / 256), 256, 0, stream>>>(obuf + (size_t)MREAL * DIM, zn);

  // QKV: [8256,768] @ [768,2304], block tile 64x256
  gemm_f16<0><<<dim3(129 * (NQKV / 256)), dim3(256), 0, stream>>>(
      x_h, wqkv_h, bqkv, qbuf, kbuf, vbuf, nullptr, NQKV, DIM, 129);

  // LayerNorm q and k (one wave per 64-wide row)
  ln_qk<<<dim3((unsigned)((long)BS * HEADS * NSEQ * 2 / 8)), dim3(256), 0, stream>>>(
      qbuf, kbuf, gq, betaq, gk, betak);

  // Flash attention
  flash_attn<<<dim3(BS * HEADS * 9), dim3(256), 0, stream>>>(qbuf, kbuf, vbuf, obuf);

  // Attention map (row 0, mean over heads)
  attn_map_kernel<<<dim3(BS), dim3(256), 0, stream>>>(qbuf, kbuf, dmap);

  // Projection: [8256,768] @ [768,768] -> d_out f32, block tile 64x256
  gemm_f16<1><<<dim3(129 * (DIM / 256)), dim3(256), 0, stream>>>(
      obuf, wproj_h, bproj, nullptr, nullptr, nullptr, out, DIM, DIM, 129);
}
